// CNNEncoder_36258113912977
// MI455X (gfx1250) — matrix operand order repaired and checked by reading, not compiled, not run on hardware
//
#include <hip/hip_runtime.h>
#include <hip/hip_bf16.h>
#include <math.h>
#include <stdint.h>

// ---------------- types ----------------
typedef __attribute__((ext_vector_type(16))) __bf16 v16bf;
typedef __attribute__((ext_vector_type(8)))  __bf16 v8bf;
typedef __attribute__((ext_vector_type(8)))  float  v8f;
typedef unsigned int u32x4 __attribute__((ext_vector_type(4)));
typedef int          i32x4 __attribute__((ext_vector_type(4)));
typedef int          i32x8 __attribute__((ext_vector_type(8)));

#define NSEQ   1024
#define LSEQ   128
#define DIM    300
#define DP     304      // DIM padded to 19 n-tiles of 16
#define NT     19       // n-tiles
#define KT1    10       // k-tiles of 32 covering 300 (padded 320)
#define TAPS   5
#define TOUT   124      // conv valid outputs
#define XROWS  132      // 128 + 4 (tap overhang), zero padded
#define XLD    320      // padded row length (bf16 elems)
#define CHB    (DP*32)  // 9728 bf16 elems per staged weight chunk (19456 B)

// 16 bf16: elems 0..7 at p, elems 8..15 at p+16 (two aligned 16B chunks)
__device__ __forceinline__ v16bf ld16c(const __bf16* p) {
    const v8bf* q = (const v8bf*)p;
    v8bf lo = q[0], hi = q[2];
    v16bf r;
#pragma unroll
    for (int i = 0; i < 8; ++i) { r[i] = lo[i]; r[8 + i] = hi[i]; }
    return r;
}
// A-fragment: elems 0..7 at p, elems 8..15 at p+16 (per 16-bit A layout)
__device__ __forceinline__ v16bf ldA(const __bf16* p) {
    const v8bf* q = (const v8bf*)p;
    v8bf lo = q[0], hi = q[2];
    v16bf r;
#pragma unroll
    for (int i = 0; i < 8; ++i) { r[i] = lo[i]; r[8 + i] = hi[i]; }
    return r;
}

// ---------------- TDM: 1D chunk copy global(bf16,19456B) -> LDS ----------------
// D# per CDNA5 ISA:
//  group0: [1:0]count=1 | [63:32]lds_addr | [120:64]global_addr | [127:126]type=2
//  group1: data_size=3 (8B); tensor_dim0=tile_dim0=tensor_dim0_stride=2432; dim1=tile1=1
__device__ __forceinline__ void tdm_chunk(const __bf16* gsrc, uint32_t lds_byte_addr) {
    uint64_t ga = (uint64_t)(uintptr_t)gsrc;
    u32x4 g0;
    g0[0] = 1u;                                   // count=1, user mode
    g0[1] = lds_byte_addr;                        // LDS destination (bytes)
    g0[2] = (uint32_t)ga;                         // global_addr[31:0]
    g0[3] = (uint32_t)((ga >> 32) & 0x01FFFFFFu)  // global_addr[56:32]
          | (2u << 30);                           // type = 2 ("image")
    i32x8 g1;
    g1[0] = (int)(3u << 16);                      // data_size = 8 bytes
    g1[1] = (int)(0x980u << 16);                  // tensor_dim0[15:0]=2432 (bits 79:64... packed [31:16])
    g1[2] = (int)(1u << 16);                      // tensor_dim0 hi=0, tensor_dim1=1
    g1[3] = (int)(0x980u << 16);                  // tensor_dim1 hi=0, tile_dim0=2432
    g1[4] = 1;                                    // tile_dim1=1, tile_dim2=0
    g1[5] = 0x980;                                // tensor_dim0_stride = 2432
    g1[6] = 0;
    g1[7] = 0;
    i32x4 z4 = {0, 0, 0, 0};
#if __clang_major__ >= 23
    i32x8 z8 = {0, 0, 0, 0, 0, 0, 0, 0};
    __builtin_amdgcn_tensor_load_to_lds(g0, g1, z4, z4, z8, 0);
#else
    __builtin_amdgcn_tensor_load_to_lds(g0, g1, z4, z4, 0);
#endif
}

// =====================================================================
// Kernel 0: one-time weight prep -> bf16, zero-padded, transposed into
// per-chunk LDS images ([kt][o 304][i 32]) so hot kernels stage via TDM.
// =====================================================================
__global__ __launch_bounds__(256)
void prep_weights(const float* __restrict__ W1, const float* __restrict__ convw,
                  const float* __restrict__ W2, const float* __restrict__ W3,
                  __bf16* __restrict__ W1T, __bf16* __restrict__ WcT,
                  __bf16* __restrict__ W2T, __bf16* __restrict__ W3T)
{
    const int t0 = blockIdx.x * blockDim.x + threadIdx.x;
    const int st = gridDim.x * blockDim.x;
    for (int x = t0; x < KT1 * CHB; x += st) {
        int kt = x / CHB, r = x % CHB, o = r >> 5, i = r & 31, k = kt * 32 + i;
        W1T[x] = (__bf16)((k < DIM && o < DIM) ? W1[k * DIM + o] : 0.0f);
    }
    for (int x = t0; x < KT1 * TAPS * CHB; x += st) {
        int c = x / CHB, r = x % CHB, o = r >> 5, i = r & 31;
        int kt = c / TAPS, tap = c % TAPS, k = kt * 32 + i;
        WcT[x] = (__bf16)((k < DIM && o < DIM) ? convw[o * (DIM * TAPS) + k * TAPS + tap] : 0.0f);
    }
    for (int x = t0; x < 2 * KT1 * CHB; x += st) {
        int c = x / CHB, r = x % CHB, o = r >> 5, i = r & 31;
        int k = (c % KT1) * 32 + i;
        int row = (c < KT1) ? k : DIM + k;
        W2T[x] = (__bf16)((k < DIM && o < DIM) ? W2[row * DIM + o] : 0.0f);
    }
    for (int x = t0; x < KT1 * CHB; x += st) {
        int kt = x / CHB, r = x % CHB, o = r >> 5, i = r & 31, k = kt * 32 + i;
        W3T[x] = (__bf16)((k < DIM && o < DIM) ? W3[k * DIM + o] : 0.0f);
    }
}

// =====================================================================
// Kernel 1: per-sequence fused  gather -> lin1 -> conv1d -> time-max
// one block/sequence; wave w owns m-tile rows 16w..16w+15 and all 19
// n-tiles; weight chunks arrive via double-buffered TDM DMA.
// =====================================================================
__global__ __launch_bounds__(256, 1)
void enc_fused_kernel(const int* __restrict__ tok,
                      const float* __restrict__ emb,
                      const float* __restrict__ b1,
                      const float* __restrict__ convb,
                      const __bf16* __restrict__ W1T,  // [10][9728]
                      const __bf16* __restrict__ WcT,  // [50][9728]
                      float* __restrict__ cnn)         // [1024][304]
{
    extern __shared__ char smem[];
    __bf16* Wbuf = (__bf16*)smem;                               // [2][9728]
    __bf16* Xbf  = (__bf16*)(smem + 2 * CHB * 2);               // [132][320]
    float*  cnnp = (float*)(smem + 2 * CHB * 2 + XROWS * XLD * 2); // [8][304]
    int*    toks = (int*)(cnnp + 8 * DP);                       // [128]
    const uint32_t ldsW = (uint32_t)(uintptr_t)Wbuf;            // LDS byte offset

    const int n    = blockIdx.x;
    const int tid  = threadIdx.x;
    const int w    = tid >> 5;
    const int lane = tid & 31;
    const int lrow = lane & 15;
    const int hi   = lane >> 4;
    const int hs   = hi ? 8 : 0;

    if (w == 0 && lane == 0) tdm_chunk(W1T, ldsW);              // prefetch chunk 0
    if (tid < LSEQ) toks[tid] = tok[n * LSEQ + tid];
    for (int i = tid; i < XROWS * XLD; i += 256) Xbf[i] = (__bf16)0.0f;

    v8f acc[NT];
#pragma unroll
    for (int t = 0; t < NT; ++t) acc[t] = (v8f){};
    __syncthreads();

    // ---------------- lin1: X1 = emb[tokens] @ W1 ----------------
    const long mytok = (long)toks[w * 16 + lrow];
    for (int kt = 0; kt < KT1; ++kt) {
        if (w == 0) {
            if (lane == 0 && kt + 1 < KT1)
                tdm_chunk(W1T + (size_t)(kt + 1) * CHB, ldsW + ((kt + 1) & 1) * CHB * 2);
            if (kt + 1 < KT1) __builtin_amdgcn_s_wait_tensorcnt(1);
            else              __builtin_amdgcn_s_wait_tensorcnt(0);
        }
        __syncthreads();
        const __bf16* Wc = Wbuf + (kt & 1) * CHB;
        // A fragment gathered directly from the embedding table
        v16bf afr;
        {
            const int c0 = kt * 32 + (hi ? 8 : 0);
#pragma unroll
            for (int i = 0; i < 8; ++i) {
                int c  = c0 + i;
                int c2 = c0 + 16 + i;
                afr[i]     = (__bf16)((c  < DIM) ? emb[mytok * DIM + c ] : 0.0f);
                afr[8 + i] = (__bf16)((c2 < DIM) ? emb[mytok * DIM + c2] : 0.0f);
            }
        }
        v16bf bfr = ld16c(&Wc[lrow * 32 + hs]);
#pragma unroll
        for (int nt = 0; nt < NT; ++nt) {
            v16bf bnx = bfr;
            if (nt + 1 < NT) bnx = ld16c(&Wc[(nt + 1) * 512 + lrow * 32 + hs]);
            acc[nt] = __builtin_amdgcn_wmma_f32_16x16x32_bf16(
                false, afr, false, bfr, (short)0, acc[nt], false, false);
            bfr = bnx;
        }
        __syncthreads();
    }

    // conv chunk 0 prefetch (buf0 free: last read was kt=8)
    if (w == 0 && lane == 0) tdm_chunk(WcT, ldsW);

    // X1 (+b1) -> bf16 into LDS
#pragma unroll
    for (int nt = 0; nt < NT; ++nt) {
        const int col = nt * 16 + lrow;
        const float bb = (col < DIM) ? b1[col] : 0.0f;
#pragma unroll
        for (int r = 0; r < 8; ++r) {
            int row = w * 16 + r + (hi ? 8 : 0);
            Xbf[row * XLD + col] = (__bf16)(acc[nt][r] + bb);
        }
    }
#pragma unroll
    for (int t = 0; t < NT; ++t) acc[t] = (v8f){};
    __syncthreads();

    // ---------------- conv1d as 50 WMMA K-chunks ----------------
    for (int c = 0; c < KT1 * TAPS; ++c) {
        if (w == 0) {
            if (lane == 0 && c + 1 < KT1 * TAPS)
                tdm_chunk(WcT + (size_t)(c + 1) * CHB, ldsW + ((c + 1) & 1) * CHB * 2);
            if (c + 1 < KT1 * TAPS) __builtin_amdgcn_s_wait_tensorcnt(1);
            else                    __builtin_amdgcn_s_wait_tensorcnt(0);
        }
        __syncthreads();
        const int kt = c / TAPS, tap = c % TAPS;
        const __bf16* Wc = Wbuf + (c & 1) * CHB;
        const int trow = w * 16 + lrow + tap;           // <= 131, zero padded
        const int c0   = kt * 32 + (hi ? 8 : 0);
        v16bf afr = ldA(&Xbf[trow * XLD + c0]);
        v16bf bfr = ld16c(&Wc[lrow * 32 + hs]);
#pragma unroll
        for (int nt = 0; nt < NT; ++nt) {
            v16bf bnx = bfr;
            if (nt + 1 < NT) bnx = ld16c(&Wc[(nt + 1) * 512 + lrow * 32 + hs]);
            acc[nt] = __builtin_amdgcn_wmma_f32_16x16x32_bf16(
                false, afr, false, bfr, (short)0, acc[nt], false, false);
            bfr = bnx;
        }
        __syncthreads();
    }

    // ---------------- max over time ----------------
#pragma unroll
    for (int nt = 0; nt < NT; ++nt) {
        float m = -INFINITY;
#pragma unroll
        for (int r = 0; r < 8; ++r) {
            int t = w * 16 + r + (hi ? 8 : 0);
            m = fmaxf(m, (t < TOUT) ? acc[nt][r] : -INFINITY);
        }
        m = fmaxf(m, __shfl_xor(m, 16, 32));
        if (hi == 0) cnnp[w * DP + nt * 16 + lrow] = m;
    }
    __syncthreads();
    for (int o = tid; o < DP; o += 256) {
        float m = cnnp[o];
#pragma unroll
        for (int ww = 1; ww < 8; ++ww) m = fmaxf(m, cnnp[ww * DP + o]);
        float cb = (o < DIM) ? convb[o] : 0.0f;
        cnn[(long)n * DP + o] = m + cb;
    }
}

// =====================================================================
// Kernel 2: head MLP  h = tanh([cnn, mention] @ W2 + b2); out = h@W3+b3
// =====================================================================
__global__ __launch_bounds__(256, 1)
void head_kernel(const float* __restrict__ cnn,    // [1024][304]
                 const float* __restrict__ men,    // [1024][300]
                 const float* __restrict__ b2,
                 const float* __restrict__ b3,
                 const __bf16* __restrict__ W2T,   // [20][9728]
                 const __bf16* __restrict__ W3T,   // [10][9728]
                 float* __restrict__ out)          // [1024][300]
{
    extern __shared__ char smem[];
    __bf16* Wbuf = (__bf16*)smem;                      // [2][9728]
    __bf16* Hbf  = (__bf16*)(smem + 2 * CHB * 2);      // [128][320]
    const uint32_t ldsW = (uint32_t)(uintptr_t)Wbuf;

    const int tid  = threadIdx.x;
    const int w    = tid >> 5;
    const int lane = tid & 31;
    const int lrow = lane & 15;
    const int hi   = lane >> 4;
    const int hs   = hi ? 8 : 0;
    const int rowbase = blockIdx.x * 128;

    if (w == 0 && lane == 0) tdm_chunk(W2T, ldsW);
    for (int i = tid; i < 128 * XLD; i += 256) Hbf[i] = (__bf16)0.0f;

    v8f acc[NT];
#pragma unroll
    for (int t = 0; t < NT; ++t) acc[t] = (v8f){};
    __syncthreads();

    const long grow = rowbase + w * 16 + lrow;

    // ---- concat @ W2 : 20 K-chunks (10 from cnn, 10 from mention) ----
    for (int c = 0; c < 2 * KT1; ++c) {
        if (w == 0) {
            if (lane == 0 && c + 1 < 2 * KT1)
                tdm_chunk(W2T + (size_t)(c + 1) * CHB, ldsW + ((c + 1) & 1) * CHB * 2);
            if (c + 1 < 2 * KT1) __builtin_amdgcn_s_wait_tensorcnt(1);
            else                 __builtin_amdgcn_s_wait_tensorcnt(0);
        }
        __syncthreads();
        const __bf16* Wc = Wbuf + (c & 1) * CHB;
        v16bf afr;
        {
            const int c0 = (c % KT1) * 32 + (hi ? 8 : 0);
#pragma unroll
            for (int i = 0; i < 8; ++i) {
                int cc = c0 + i, c2 = c0 + 16 + i;
                float v0, v1;
                if (c < KT1) {
                    v0 = (cc < DP) ? cnn[grow * DP + cc] : 0.0f;
                    v1 = (c2 < DP) ? cnn[grow * DP + c2] : 0.0f;
                } else {
                    v0 = (cc < DIM) ? men[grow * DIM + cc] : 0.0f;
                    v1 = (c2 < DIM) ? men[grow * DIM + c2] : 0.0f;
                }
                afr[i]     = (__bf16)v0;
                afr[8 + i] = (__bf16)v1;
            }
        }
        v16bf bfr = ld16c(&Wc[lrow * 32 + hs]);
#pragma unroll
        for (int nt = 0; nt < NT; ++nt) {
            v16bf bnx = bfr;
            if (nt + 1 < NT) bnx = ld16c(&Wc[(nt + 1) * 512 + lrow * 32 + hs]);
            acc[nt] = __builtin_amdgcn_wmma_f32_16x16x32_bf16(
                false, afr, false, bfr, (short)0, acc[nt], false, false);
            bfr = bnx;
        }
        __syncthreads();
    }

    if (w == 0 && lane == 0) tdm_chunk(W3T, ldsW);   // W3 chunk 0 into buf0

    // h = tanh(.+b2) -> bf16 LDS
#pragma unroll
    for (int nt = 0; nt < NT; ++nt) {
        const int col = nt * 16 + lrow;
        const float bb = (col < DIM) ? b2[col] : 0.0f;
#pragma unroll
        for (int r = 0; r < 8; ++r) {
            int rl = w * 16 + r + (hi ? 8 : 0);
            Hbf[rl * XLD + col] = (__bf16)tanhf(acc[nt][r] + bb);
        }
    }
#pragma unroll
    for (int t = 0; t < NT; ++t) acc[t] = (v8f){};
    __syncthreads();

    // ---- h @ W3 ----
    for (int c = 0; c < KT1; ++c) {
        if (w == 0) {
            if (lane == 0 && c + 1 < KT1)
                tdm_chunk(W3T + (size_t)(c + 1) * CHB, ldsW + ((c + 1) & 1) * CHB * 2);
            if (c + 1 < KT1) __builtin_amdgcn_s_wait_tensorcnt(1);
            else             __builtin_amdgcn_s_wait_tensorcnt(0);
        }
        __syncthreads();
        const __bf16* Wc = Wbuf + (c & 1) * CHB;
        const int c0 = c * 32 + (hi ? 8 : 0);
        v16bf afr = ldA(&Hbf[(w * 16 + lrow) * XLD + c0]);
        v16bf bfr = ld16c(&Wc[lrow * 32 + hs]);
#pragma unroll
        for (int nt = 0; nt < NT; ++nt) {
            v16bf bnx = bfr;
            if (nt + 1 < NT) bnx = ld16c(&Wc[(nt + 1) * 512 + lrow * 32 + hs]);
            acc[nt] = __builtin_amdgcn_wmma_f32_16x16x32_bf16(
                false, afr, false, bfr, (short)0, acc[nt], false, false);
            bfr = bnx;
        }
        __syncthreads();
    }
    // out = acc + b3
#pragma unroll
    for (int nt = 0; nt < NT; ++nt) {
        const int col = nt * 16 + lrow;
        if (col < DIM) {
            const float bb = b3[col];
#pragma unroll
            for (int r = 0; r < 8; ++r) {
                long row = rowbase + w * 16 + r + (hi ? 8 : 0);
                out[row * DIM + col] = acc[nt][r] + bb;
            }
        }
    }
}

// =====================================================================
extern "C" void kernel_launch(void* const* d_in, const int* in_sizes, int n_in,
                              void* d_out, int out_size, void* d_ws, size_t ws_size,
                              hipStream_t stream) {
    const int*   tok   = (const int*)  d_in[0];
    const float* men   = (const float*)d_in[1];
    const float* emb   = (const float*)d_in[2];
    const float* W1    = (const float*)d_in[3];
    const float* b1    = (const float*)d_in[4];
    const float* convw = (const float*)d_in[5];
    const float* convb = (const float*)d_in[6];
    const float* W2    = (const float*)d_in[7];
    const float* b2    = (const float*)d_in[8];
    const float* W3    = (const float*)d_in[9];
    const float* b3    = (const float*)d_in[10];
    float* outp = (float*)d_out;

    // workspace layout
    char* ws = (char*)d_ws;
    float*  cnn = (float*)ws;                               // [1024][304] f32
    size_t  off = (size_t)NSEQ * DP * sizeof(float);
    __bf16* W1T = (__bf16*)(ws + off); off += (size_t)KT1 * CHB * 2;
    __bf16* WcT = (__bf16*)(ws + off); off += (size_t)KT1 * TAPS * CHB * 2;
    __bf16* W2T = (__bf16*)(ws + off); off += (size_t)2 * KT1 * CHB * 2;
    __bf16* W3T = (__bf16*)(ws + off);

    const int smem1 = 2 * CHB * 2 + XROWS * XLD * 2 + 8 * DP * 4 + LSEQ * 4;
    const int smem2 = 2 * CHB * 2 + 128 * XLD * 2;
    hipFuncSetAttribute((const void*)enc_fused_kernel,
                        hipFuncAttributeMaxDynamicSharedMemorySize, smem1);
    hipFuncSetAttribute((const void*)head_kernel,
                        hipFuncAttributeMaxDynamicSharedMemorySize, smem2);

    prep_weights<<<512, 256, 0, stream>>>(W1, convw, W2, W3, W1T, WcT, W2T, W3T);
    enc_fused_kernel<<<NSEQ, 256, smem1, stream>>>(tok, emb, b1, convb, W1T, WcT, cnn);
    head_kernel<<<NSEQ / 128, 256, smem2, stream>>>(cnn, men, b2, b3, W2T, W3T, outp);
}
